// AsyncTFBase_29222957482461
// MI455X (gfx1250) — matrix operand order repaired and checked by reading, not compiled, not run on hardware
//
#include <hip/hip_runtime.h>

// ---------------- problem constants ----------------
#define T_DIM 32
#define B_DIM 64
#define M_DIM 2048      // T*B
#define D_DIM 2048
#define H_DIM 1000
#define H_PAD 1024
#define R_LR  5

typedef __attribute__((ext_vector_type(16))) __bf16 v16bf;
typedef __attribute__((ext_vector_type(8)))  float  v8f;
typedef __attribute__((ext_vector_type(4)))  int    v4i_t;

union Frag32 { uint4 u[2]; v16bf v; };

// CDNA5 async global->LDS path (ASYNCcnt); falls back to sync loads if the
// builtins are not declared by this toolchain.
// Probe-discovered signature: (v4i addrspace(1)* src, v4i addrspace(3)* dst, Ii, Ii)
#if defined(__AMDGCN__) && __has_builtin(__builtin_amdgcn_global_load_async_to_lds_b128) && \
    __has_builtin(__builtin_amdgcn_s_wait_asynccnt)
#define USE_ASYNC_LDS 1
typedef __attribute__((address_space(1))) v4i_t glb_v4i;
typedef __attribute__((address_space(3))) v4i_t lds_v4i;
#else
#define USE_ASYNC_LDS 0
#endif

// ---------------- conversion kernels ----------------
__global__ __launch_bounds__(256) void cvt_f32_bf16(const float* __restrict__ in,
                                                    __bf16* __restrict__ out, int n) {
  int i = blockIdx.x * 256 + threadIdx.x;
  if (i < n) out[i] = (__bf16)in[i];
}

// W[K][N] f32 -> out[Kpad][Npad] bf16 (zero padded)
__global__ __launch_bounds__(256) void cvt_pad_w(const float* __restrict__ W,
                                                 __bf16* __restrict__ out,
                                                 int K, int N, int Kpad, int Npad) {
  int i = blockIdx.x * 256 + threadIdx.x;
  int tot = Kpad * Npad;
  if (i >= tot) return;
  int n = i % Npad, k = i / Npad;
  float v = (k < K && n < N) ? W[(size_t)k * N + n] : 0.f;
  out[i] = (__bf16)v;
}

// ---------------- WMMA bf16 GEMM ----------------
// Block tile 128x64, 8 wave32s, each wave owns a 32x32 C patch (4 WMMA/K-step).
// A tile (128x32) is DMA'd global->LDS asynchronously (triple buffered);
// B tile (32x64) goes through VGPRs and is stored transposed [n][k] (double
// buffered). One barrier per K-step.
#define LDH 40  // LDS row stride in halfs (32 + 8 pad) -> 80B rows, 16B aligned

__global__ __launch_bounds__(256)
void gemm_wmma_bf16(const __bf16* __restrict__ A, int lda,
                    const __bf16* __restrict__ Bw, int ldb, int Npad,
                    const float* __restrict__ bias,
                    float* __restrict__ Cf, __bf16* __restrict__ Cb,
                    int ldc, int N, int Nstore, int ksteps, int relu) {
  __shared__ __bf16 lA[3][128 * LDH];
  __shared__ __bf16 lB[2][64 * LDH];

  const int tid  = threadIdx.x;
  const int lane = tid & 31;
  const int wv   = tid >> 5;
  const int m0   = blockIdx.x * 128;
  const int n0   = blockIdx.y * 64;
  const int mrow0 = (wv & 3) * 32;   // wave's 32-row stripe
  const int nc0   = (wv >> 2) * 32;  // wave's 32-col stripe

  v8f c00 = {}, c01 = {}, c10 = {}, c11 = {};

  // cooperative-load index precompute
  const int a_row = tid >> 2;        // 0..63 (+64 for second chunk)
  const int a_col = (tid & 3) * 8;   // 0,8,16,24
  const int b_k   = tid >> 3;        // 0..31
  const int b_n   = (tid & 7) * 8;   // 0..56
  const bool b_in = (n0 + b_n) < Npad;  // Npad mult of 8 -> group all-in/out

  // ---- tile movers ----
  auto loadB = [&](int ks) -> uint4 {
    uint4 d = {0u, 0u, 0u, 0u};
    if (b_in)
      d = *(const uint4*)(Bw + (size_t)(ks * 32 + b_k) * ldb + n0 + b_n);
    return d;
  };
  auto storeB = [&](uint4 d, int buf) {
    const __bf16* h = (const __bf16*)&d;
#pragma unroll
    for (int j = 0; j < 8; ++j)
      lB[buf][(b_n + j) * LDH + b_k] = h[j];
  };

#if USE_ASYNC_LDS
  auto issueA = [&](int ks, int buf) {
#pragma unroll
    for (int i = 0; i < 2; ++i) {
      const __bf16* g = A + (size_t)(m0 + a_row + i * 64) * lda + ks * 32 + a_col;
      __bf16* l = &lA[buf][(a_row + i * 64) * LDH + a_col];
      __builtin_amdgcn_global_load_async_to_lds_b128((glb_v4i*)(void*)g,
                                                     (lds_v4i*)(void*)l, 0, 0);
    }
  };
#else
  uint4 aR0, aR1;
  auto loadA = [&](int ks) {
    aR0 = *(const uint4*)(A + (size_t)(m0 + a_row) * lda + ks * 32 + a_col);
    aR1 = *(const uint4*)(A + (size_t)(m0 + a_row + 64) * lda + ks * 32 + a_col);
  };
  auto storeA = [&](int buf) {
    *(uint4*)&lA[buf][a_row * LDH + a_col] = aR0;
    *(uint4*)&lA[buf][(a_row + 64) * LDH + a_col] = aR1;
  };
#endif

  // ---- prologue: tile 0 in flight ----
#if USE_ASYNC_LDS
  issueA(0, 0);
#else
  loadA(0);
#endif
  uint4 gB = loadB(0);

  for (int ks = 0; ks < ksteps; ++ks) {
    const int acur = ks % 3;
    const int bcur = ks & 1;

#if USE_ASYNC_LDS
    storeB(gB, bcur);
    if (ks + 1 < ksteps) {
      issueA(ks + 1, (ks + 1) % 3);
      gB = loadB(ks + 1);
      __builtin_amdgcn_s_wait_asynccnt(2);  // tile ks landed, ks+1 in flight
    } else {
      __builtin_amdgcn_s_wait_asynccnt(0);
    }
#else
    storeA(acur);
    storeB(gB, bcur);
    if (ks + 1 < ksteps) {
      loadA(ks + 1);
      gB = loadB(ks + 1);
    }
#endif
    __syncthreads();  // single barrier per K-step (multi-buffered LDS)

    // ---- fragments per ISA 7.12.2 lane layouts ----
    Frag32 fa0, fa1, fb0, fb1;
    {
      // A 16x32: lanes 0-15 (M=lane) K 0-7 & 16-23; lanes 16-31 K 8-15 & 24-31
      const __bf16* ap = &lA[acur][(mrow0 + (lane & 15)) * LDH + ((lane >> 4) * 8)];
      fa0.u[0] = *(const uint4*)ap;
      fa0.u[1] = *(const uint4*)(ap + 16);
      const __bf16* aq = ap + 16 * LDH;
      fa1.u[0] = *(const uint4*)aq;
      fa1.u[1] = *(const uint4*)(aq + 16);
    }
    {
      // B 32x16: lanes 0-15 (N=lane) K 0-7 & 16-23; lanes 16-31 K 8-15 & 24-31
      const __bf16* bp = &lB[bcur][(nc0 + (lane & 15)) * LDH + ((lane >> 4) * 8)];
      fb0.u[0] = *(const uint4*)bp;
      fb0.u[1] = *(const uint4*)(bp + 16);
      const __bf16* bq = bp + 16 * LDH;
      fb1.u[0] = *(const uint4*)bq;
      fb1.u[1] = *(const uint4*)(bq + 16);
    }

    c00 = __builtin_amdgcn_wmma_f32_16x16x32_bf16(false, fa0.v, false, fb0.v,
                                                  (short)0, c00, false, false);
    c01 = __builtin_amdgcn_wmma_f32_16x16x32_bf16(false, fa0.v, false, fb1.v,
                                                  (short)0, c01, false, false);
    c10 = __builtin_amdgcn_wmma_f32_16x16x32_bf16(false, fa1.v, false, fb0.v,
                                                  (short)0, c10, false, false);
    c11 = __builtin_amdgcn_wmma_f32_16x16x32_bf16(false, fa1.v, false, fb1.v,
                                                  (short)0, c11, false, false);
    // no trailing barrier: next iteration writes different LDS buffers
  }

  // ---- epilogue: C layout VGPR r -> M = sub*16 + r + (lane>=16)*8 ----
  const int gcolbase = n0 + nc0 + (lane & 15);
  const int growbase = m0 + mrow0 + ((lane >> 4) * 8);
#pragma unroll
  for (int i = 0; i < 2; ++i) {
#pragma unroll
    for (int r = 0; r < 8; ++r) {
      const int grow = growbase + i * 16 + r;
#pragma unroll
      for (int s = 0; s < 2; ++s) {
        const int gcol = gcolbase + s * 16;
        if (gcol >= Nstore) continue;
        float v;
        if (i == 0) v = s ? c01[r] : c00[r];
        else        v = s ? c11[r] : c10[r];
        if (gcol < N) {
          if (bias) v += bias[gcol];
          if (relu && v < 0.f) v = 0.f;
        } else {
          v = 0.f;  // zero the pad columns of hidden buffers
        }
        const size_t off = (size_t)grow * ldc + gcol;
        if (Cf) Cf[off] = v;
        else    Cb[off] = (__bf16)v;
      }
    }
  }
}

// ---------------- rank-R bilinear contraction ----------------
// out[t,b,i,j] = sum_r a[t',b,i,r] * bm[t,b,r,j]; t' = temporal ? max(t-1,0) : t
__global__ __launch_bounds__(256)
void bilinear_kernel(const float* __restrict__ Af, const float* __restrict__ Bf,
                     float* __restrict__ Out, int ca, int cb, int temporal, int total) {
  int idx = blockIdx.x * 256 + threadIdx.x;
  if (idx >= total) return;
  int j = idx % cb;
  int t2 = idx / cb;
  int i = t2 % ca;
  t2 /= ca;
  int b = t2 % B_DIM;
  int t = t2 / B_DIM;
  int ta = temporal ? (t == 0 ? 0 : t - 1) : t;
  const float* ap = Af + ((size_t)(ta * B_DIM + b) * ca + i) * R_LR;
  const float* bp = Bf + ((size_t)(t * B_DIM + b) * R_LR) * cb + j;
  float s = 0.f;
#pragma unroll
  for (int r = 0; r < R_LR; ++r) s += ap[r] * bp[(size_t)r * cb];
  Out[idx] = s;
}

// ---------------- host orchestration ----------------
extern "C" void kernel_launch(void* const* d_in, const int* in_sizes, int n_in,
                              void* d_out, int out_size, void* d_ws, size_t ws_size,
                              hipStream_t stream) {
  (void)in_sizes; (void)n_in; (void)out_size; (void)ws_size;

  const float* f   = (const float*)d_in[0];
  const float* s1W = (const float*)d_in[1];  const float* s1b = (const float*)d_in[2];
  const float* s2W = (const float*)d_in[3];  const float* s2b = (const float*)d_in[4];
  const float* s3W = (const float*)d_in[5];  const float* s3b = (const float*)d_in[6];
  const float* oW  = (const float*)d_in[7];  const float* ob  = (const float*)d_in[8];
  const float* vW  = (const float*)d_in[9];  const float* vb  = (const float*)d_in[10];

  // workspace layout (bytes)
  char* ws = (char*)d_ws;
  __bf16* f16  = (__bf16*)(ws);                               //  8,388,608
  __bf16* wbuf = (__bf16*)(ws + 8u * 1024 * 1024 + 196608);   //  4,194,304 max use
  __bf16* h1   = (__bf16*)(ws + 12u * 1024 * 1024 + 196608);  //  2048*1024 bf16
  __bf16* h2   = (__bf16*)(ws + 16u * 1024 * 1024 + 196608);
  char*   facb = ws + 20u * 1024 * 1024 + 196608;

  static const int fac_out[24] = {80, 190, 190, 165, 165, 80, 80, 80,
                                  190, 190, 165, 165, 80, 190, 190, 165,
                                  165, 80, 190, 80, 165, 190, 80, 165};
  float* fac[24];
  {
    size_t off = 0;
    for (int i = 0; i < 24; ++i) {
      fac[i] = (float*)(facb + off);
      off += (size_t)M_DIM * fac_out[i] * sizeof(float);  // multiples of 8KB
    }
  }

  auto cvtW = [&](const float* W, int K, int N, int Kpad, int Npad) {
    int tot = Kpad * Npad;
    cvt_pad_w<<<dim3((tot + 255) / 256), dim3(256), 0, stream>>>(W, wbuf, K, N, Kpad, Npad);
  };
  auto gemm = [&](const __bf16* A, int lda, int Kpad, int ldb, int Npad,
                  const float* bias, float* Cf, __bf16* Cb,
                  int ldc, int N, int Nstore, int relu) {
    dim3 g(M_DIM / 128, (Nstore + 63) / 64);
    gemm_wmma_bf16<<<g, dim3(256), 0, stream>>>(A, lda, wbuf, ldb, Npad, bias,
                                                Cf, Cb, ldc, N, Nstore, Kpad / 32, relu);
  };

  // 0) rgb_feat -> bf16
  cvt_f32_bf16<<<dim3((M_DIM * D_DIM + 255) / 256), dim3(256), 0, stream>>>(
      f, f16, M_DIM * D_DIM);

  // output offsets (floats), tuple order
  static const size_t osz[15] = {
      (size_t)M_DIM * 16,   (size_t)M_DIM * 38,   (size_t)M_DIM * 33,
      (size_t)M_DIM * 608,  (size_t)M_DIM * 1254, (size_t)M_DIM * 528,
      (size_t)M_DIM * 256,  (size_t)M_DIM * 1444, (size_t)M_DIM * 1089,
      (size_t)M_DIM * 608,  (size_t)M_DIM * 1254, (size_t)M_DIM * 528,
      (size_t)M_DIM * 608,  (size_t)M_DIM * 1254, (size_t)M_DIM * 528};
  size_t ooff[15];
  {
    size_t a = 0;
    for (int i = 0; i < 15; ++i) { ooff[i] = a; a += osz[i]; }
  }
  float* outp = (float*)d_out;

  // 1) s path: D->H relu, H->H relu, H->S
  cvtW(s1W, D_DIM, H_DIM, D_DIM, H_DIM);
  gemm(f16, D_DIM, D_DIM, H_DIM, H_DIM, s1b, nullptr, h1, H_PAD, H_DIM, H_PAD, 1);
  cvtW(s2W, H_DIM, H_DIM, H_PAD, H_DIM);
  gemm(h1, H_PAD, H_PAD, H_DIM, H_DIM, s2b, nullptr, h2, H_PAD, H_DIM, H_PAD, 1);
  cvtW(s3W, H_DIM, 16, H_PAD, 16);
  gemm(h2, H_PAD, H_PAD, 16, 16, s3b, outp + ooff[0], nullptr, 16, 16, 16, 0);

  // 2) o, v : D -> 38 / 33
  cvtW(oW, D_DIM, 38, D_DIM, 40);
  gemm(f16, D_DIM, D_DIM, 40, 40, ob, outp + ooff[1], nullptr, 38, 38, 38, 0);
  cvtW(vW, D_DIM, 33, D_DIM, 40);
  gemm(f16, D_DIM, D_DIM, 40, 40, vb, outp + ooff[2], nullptr, 33, 33, 33, 0);

  // 3) 24 heads: (D->H relu) then (H->out)
  for (int i = 0; i < 24; ++i) {
    const float* W1 = (const float*)d_in[11 + 4 * i];
    const float* b1 = (const float*)d_in[12 + 4 * i];
    const float* W2 = (const float*)d_in[13 + 4 * i];
    const float* b2 = (const float*)d_in[14 + 4 * i];
    const int no = fac_out[i];
    const int npad = (no + 7) & ~7;
    cvtW(W1, D_DIM, H_DIM, D_DIM, H_DIM);
    gemm(f16, D_DIM, D_DIM, H_DIM, H_DIM, b1, nullptr, h1, H_PAD, H_DIM, H_PAD, 1);
    cvtW(W2, H_DIM, no, H_PAD, npad);
    gemm(h1, H_PAD, H_PAD, npad, npad, b2, fac[i], nullptr, no, no, no, 0);
  }

  // 4) bilinear contractions
  struct Pair { int ca, cb, temporal, oi; };
  static const Pair pairs[12] = {
      {16, 38, 0, 3},  {38, 33, 0, 4},  {33, 16, 0, 5},
      {16, 16, 1, 6},  {38, 38, 1, 7},  {33, 33, 1, 8},
      {16, 38, 1, 9},  {38, 33, 1, 10}, {33, 16, 1, 11},
      {38, 16, 1, 12}, {33, 38, 1, 13}, {16, 33, 1, 14}};
  for (int p = 0; p < 12; ++p) {
    const Pair pr = pairs[p];
    int total = M_DIM * pr.ca * pr.cb;
    bilinear_kernel<<<dim3((total + 255) / 256), dim3(256), 0, stream>>>(
        fac[2 * p], fac[2 * p + 1], outp + ooff[pr.oi], pr.ca, pr.cb, pr.temporal, total);
  }
}
